// TopkGAT_29334626631944
// MI455X (gfx1250) — matrix operand order repaired and checked by reading, not compiled, not run on hardware
//
#include <hip/hip_runtime.h>
#include <hip/hip_bf16.h>
#include <math.h>

// ---------------------------------------------------------------------------
// TopkGAT forward for MI455X (gfx1250, wave32, WMMA).
//
// Input layout assumption (setup_inputs() dict insertion order, leaves flat):
//   d_in[0] = x (N*128 f32), d_in[1] = edge_index (2*E i32), d_in[2] = batch
//   per layer i=0..3 (11 tensors from d_in[3]):
//     conv{i}: Wl bl Wr br att bias | bn{i}: gamma beta mean var | pool{i}: w
//   d_in[47..48]=dense1.W,b  d_in[49..50]=dense2.W,b
// Output: d_out = (128 x 10) f32 logits.
// ---------------------------------------------------------------------------

#define NN_   65536
#define NPG_  512
#define BB_   128
#define EE_   1048576
#define HH_   128
#define FIN_  128
#define CC_   10
#define FLATD_ 1024
#define HID_  512
#define NEGV  (-1e30f)

typedef __attribute__((ext_vector_type(16))) __bf16 v16bf;
typedef __attribute__((ext_vector_type(8)))  __bf16 v8bf;
typedef __attribute__((ext_vector_type(8)))  float  v8f;

// ---------------------------------------------------------------------------
// A pack: f32 -> bf16, contiguous (row-major), dst[i] = bf16(src[i]).
// Vectorized: each thread converts 8 floats -> 4 dwords of packed bf16.
// ---------------------------------------------------------------------------
__global__ void pack_a(const float* __restrict__ src, __bf16* __restrict__ dst, int nelem)
{
    const int i = (blockIdx.x * blockDim.x + threadIdx.x) * 8;
    if (i >= nelem) return;
#pragma unroll
    for (int j = 0; j < 8; ++j) dst[i + j] = (__bf16)src[i + j];
}

// ---------------------------------------------------------------------------
// B pack: fragment-ordered bf16.  dst[((k>>4)*Nn + n)*16 + (k&15)] = B[k][n].
// A lane's WMMA B-fragment (rows k0+half*16..+15 of one column) is then
// 32 contiguous bytes -> two global_load_b128 per fragment.
// ---------------------------------------------------------------------------
__global__ void pack_b_frag(const float* __restrict__ src, __bf16* __restrict__ dst,
                            int K, int Nn)
{
    const int i = blockIdx.x * blockDim.x + threadIdx.x;
    if (i >= K * Nn) return;
    const int k = i / Nn;
    const int n = i - k * Nn;
    dst[((size_t)(k >> 4) * Nn + n) * 16 + (k & 15)] = (__bf16)src[i];
}

// ---------------------------------------------------------------------------
// GEMM: C[M,Nn] = relu?( A[M,K] @ B[K,Nn] + bias[Nn] ), bf16 WMMA 16x16x32.
// Ap: bf16 row-major.  Bp: fragment-ordered (see pack_b_frag).
// One wave per (16 x 16*TNT) tile group; A fragment reused across TNT wmmas.
// Lane layouts used here (wave32, contiguous k; A and B agree):
//   A 16x32: lane L row m=L%16; elems e -> K=k0+(L/16)*16+e
//   B 32x16: lane L col n=L%16; elems e -> K=k0+(L/16)*16+e
//   C/D:     lane L col n=L%16; elem r -> row r+8*(L/16)
// Out-of-range columns are clamped on load (divergence-free), masked on store.
// ---------------------------------------------------------------------------
template <int TNT, int RELU>
__global__ void __launch_bounds__(256)
wmma_gemm(const __bf16* __restrict__ Ap, const __bf16* __restrict__ Bp,
          const float* __restrict__ bias, float* __restrict__ C,
          int M, int K, int Nn)
{
    const int wave   = (blockIdx.x * blockDim.x + threadIdx.x) >> 5;
    const int nwaves = (gridDim.x * blockDim.x) >> 5;
    const int lane   = threadIdx.x & 31;
    const int half   = lane >> 4;
    const int l16    = lane & 15;

    const int tilesN  = (Nn + 15) >> 4;
    const int groupsN = (tilesN + TNT - 1) / TNT;
    const int tilesM  = M >> 4;
    const int total   = tilesM * groupsN;

    for (int g = wave; g < total; g += nwaves) {
        const int tm  = g / groupsN;
        const int gn  = g - tm * groupsN;
        const int tn0 = gn * TNT;

        int col[TNT];
#pragma unroll
        for (int t = 0; t < TNT; ++t) {
            const int cb = (tn0 + t) * 16 + l16;
            col[t] = (cb < Nn) ? cb : (Nn - 1);   // clamp -> unconditional loads
        }
        v8f c[TNT];
#pragma unroll
        for (int t = 0; t < TNT; ++t) c[t] = (v8f){};

        const __bf16* arow = Ap + (size_t)(tm * 16 + l16) * K;

        for (int k0 = 0; k0 < K; k0 += 32) {
            const v8bf alo = *(const v8bf*)(arow + k0 + half * 16);
            const v8bf ahi = *(const v8bf*)(arow + k0 + half * 16 + 8);
            const v16bf a = __builtin_shufflevector(alo, ahi,
                0, 1, 2, 3, 4, 5, 6, 7, 8, 9, 10, 11, 12, 13, 14, 15);
#pragma unroll
            for (int t = 0; t < TNT; ++t) {
                const __bf16* bp =
                    Bp + (((size_t)((k0 >> 4) + half) * Nn + col[t]) << 4);
                const v8bf b0 = *(const v8bf*)bp;
                const v8bf b1 = *(const v8bf*)(bp + 8);
                const v16bf b = __builtin_shufflevector(b0, b1,
                    0, 1, 2, 3, 4, 5, 6, 7, 8, 9, 10, 11, 12, 13, 14, 15);
                c[t] = __builtin_amdgcn_wmma_f32_16x16x32_bf16(
                           false, a, false, b, (short)0, c[t], false, false);
            }
        }

#pragma unroll
        for (int t = 0; t < TNT; ++t) {
            const int cb = (tn0 + t) * 16 + l16;
            if (cb < Nn) {
                const float bv = bias ? bias[cb] : 0.f;
#pragma unroll
                for (int r = 0; r < 8; ++r) {
                    float v = c[t][r] + bv;
                    if (RELU) v = fmaxf(v, 0.f);
                    C[(size_t)(tm * 16 + r + 8 * half) * Nn + cb] = v;
                }
            }
        }
    }
}

// ---------------------------------------------------------------------------
// Elementwise / edge kernels
// ---------------------------------------------------------------------------
__device__ __forceinline__ void atomicMaxF(float* addr, float val)
{
    if (val >= 0.f) atomicMax((int*)addr, __float_as_int(val));
    else            atomicMin((unsigned int*)addr, __float_as_uint(val));
}

__global__ void init_masks(unsigned char* nmask, unsigned char* emask)
{
    int i = blockIdx.x * blockDim.x + threadIdx.x;
    if (i < EE_) emask[i] = 1;
    if (i < NN_) nmask[i] = 1;
}

__global__ void init_layer(float* acc, float* emax, float* den)
{
    const int i = blockIdx.x * blockDim.x + threadIdx.x;
    if (i < NN_ * HH_ / 4) ((float4*)acc)[i] = make_float4(0.f, 0.f, 0.f, 0.f);
    if (i < NN_) { emax[i] = NEGV; den[i] = 0.f; }
}

// e = leaky_relu(xl[s]+xr[d], 0.2) . att  (or NEG if masked); wave per edge,
// float4 per lane (128 feats / 32 lanes).
__global__ void edge_score(const float* __restrict__ xl, const float* __restrict__ xr,
                           const float* __restrict__ att,
                           const int* __restrict__ src, const int* __restrict__ dst,
                           const unsigned char* __restrict__ emask,
                           const unsigned char* __restrict__ nmask,
                           float* __restrict__ evals)
{
    const int w    = (blockIdx.x * blockDim.x + threadIdx.x) >> 5;
    const int lane = threadIdx.x & 31;
    if (w >= EE_ + NN_) return;
    int s, d; bool m;
    if (w < EE_) { s = src[w]; d = dst[w]; m = emask[w] != 0; }
    else         { s = d = w - EE_;        m = nmask[s] != 0; }
    float acc = 0.f;
    if (m) {
        const int f = lane * 4;
        const float4 a4 = *(const float4*)(xl + (size_t)s * HH_ + f);
        const float4 b4 = *(const float4*)(xr + (size_t)d * HH_ + f);
        const float4 w4 = *(const float4*)(att + f);
        float v;
        v = a4.x + b4.x; acc += (v > 0.f ? v : 0.2f * v) * w4.x;
        v = a4.y + b4.y; acc += (v > 0.f ? v : 0.2f * v) * w4.y;
        v = a4.z + b4.z; acc += (v > 0.f ? v : 0.2f * v) * w4.z;
        v = a4.w + b4.w; acc += (v > 0.f ? v : 0.2f * v) * w4.w;
#pragma unroll
        for (int off = 16; off > 0; off >>= 1) acc += __shfl_down(acc, off, 32);
    }
    if (lane == 0) evals[w] = m ? acc : NEGV;
}

__global__ void edge_max(const float* __restrict__ evals,
                         const int* __restrict__ dst, float* __restrict__ emax)
{
    const int i = blockIdx.x * blockDim.x + threadIdx.x;
    if (i >= EE_ + NN_) return;
    const int d = (i < EE_) ? dst[i] : (i - EE_);
    atomicMaxF(&emax[d], evals[i]);
}

__global__ void edge_expsum(float* __restrict__ evals,
                            const int* __restrict__ src, const int* __restrict__ dst,
                            const unsigned char* __restrict__ emask,
                            const unsigned char* __restrict__ nmask,
                            const float* __restrict__ emax, float* __restrict__ den)
{
    const int i = blockIdx.x * blockDim.x + threadIdx.x;
    if (i >= EE_ + NN_) return;
    int d; bool m;
    if (i < EE_) { d = dst[i]; m = emask[i] != 0; }
    else         { d = i - EE_; m = nmask[d] != 0; }
    float ex = 0.f;
    if (m) ex = expf(evals[i] - emax[d]);
    evals[i] = ex;
    if (ex != 0.f) atomicAdd(&den[d], ex);
}

// acc[d,:] += (ex/(den[d]+eps)) * xl[s,:] ; wave per edge, float4 gather,
// 4 scalar f32 atomics per lane (no vector atomic add).
__global__ void edge_scatter(const float* __restrict__ evals,
                             const int* __restrict__ src, const int* __restrict__ dst,
                             const float* __restrict__ den,
                             const float* __restrict__ xl, float* __restrict__ acc)
{
    const int w    = (blockIdx.x * blockDim.x + threadIdx.x) >> 5;
    const int lane = threadIdx.x & 31;
    if (w >= EE_ + NN_) return;
    int s, d;
    if (w < EE_) { s = src[w]; d = dst[w]; }
    else         { s = d = w - EE_; }
    const float ex = evals[w];
    if (ex == 0.f) return;
    const float alpha = ex / (den[d] + 1e-16f);
    const int f = lane * 4;
    const float4 v4 = *(const float4*)(xl + (size_t)s * HH_ + f);
    float* a = acc + (size_t)d * HH_ + f;
    atomicAdd(a + 0, alpha * v4.x);
    atomicAdd(a + 1, alpha * v4.y);
    atomicAdd(a + 2, alpha * v4.z);
    atomicAdd(a + 3, alpha * v4.w);
}

// h = relu( bn(acc + conv_bias) ) * nmask ; float4 per thread.
__global__ void bias_bn_relu(const float* __restrict__ acc,
                             const float* __restrict__ cbias,
                             const float* __restrict__ gamma, const float* __restrict__ beta,
                             const float* __restrict__ mean,  const float* __restrict__ var,
                             const unsigned char* __restrict__ nmask,
                             float* __restrict__ h)
{
    const int i = blockIdx.x * blockDim.x + threadIdx.x;   // over N*H/4
    if (i >= NN_ * HH_ / 4) return;
    const int e = i * 4;
    const int f = e & (HH_ - 1);
    const int n = e >> 7;
    const float4 a4 = ((const float4*)acc)[i];
    const float4 cb = *(const float4*)(cbias + f);
    const float4 g4 = *(const float4*)(gamma + f);
    const float4 be = *(const float4*)(beta + f);
    const float4 mn = *(const float4*)(mean + f);
    const float4 vr = *(const float4*)(var + f);
    const float msk = nmask[n] ? 1.f : 0.f;
    float4 o;
    o.x = fmaxf((a4.x + cb.x - mn.x) * (g4.x * rsqrtf(vr.x + 1e-5f)) + be.x, 0.f) * msk;
    o.y = fmaxf((a4.y + cb.y - mn.y) * (g4.y * rsqrtf(vr.y + 1e-5f)) + be.y, 0.f) * msk;
    o.z = fmaxf((a4.z + cb.z - mn.z) * (g4.z * rsqrtf(vr.z + 1e-5f)) + be.z, 0.f) * msk;
    o.w = fmaxf((a4.w + cb.w - mn.w) * (g4.w * rsqrtf(vr.w + 1e-5f)) + be.w, 0.f) * msk;
    ((float4*)h)[i] = o;
}

// score[n] = h[n,:].w / (||w||+eps) ; wave per node, float4 per lane.
__global__ void score_kernel(const float* __restrict__ h, const float* __restrict__ w,
                             float* __restrict__ score)
{
    const int n    = (blockIdx.x * blockDim.x + threadIdx.x) >> 5;
    const int lane = threadIdx.x & 31;
    if (n >= NN_) return;
    const int f = lane * 4;
    const float4 h4 = *(const float4*)(h + (size_t)n * HH_ + f);
    const float4 w4 = *(const float4*)(w + f);
    float dot = h4.x * w4.x + h4.y * w4.y + h4.z * w4.z + h4.w * w4.w;
    float nrm = w4.x * w4.x + w4.y * w4.y + w4.z * w4.z + w4.w * w4.w;
#pragma unroll
    for (int off = 16; off > 0; off >>= 1) {
        dot += __shfl_down(dot, off, 32);
        nrm += __shfl_down(nrm, off, 32);
    }
    if (lane == 0) score[n] = dot / (sqrtf(nrm) + 1e-16f);
}

// Per-graph exact top-k by rank counting (tie: lower index wins, like top_k).
__global__ void __launch_bounds__(NPG_) topk_kernel(const float* __restrict__ score,
                                                    unsigned char* __restrict__ nmask,
                                                    float* __restrict__ gate, int k)
{
    __shared__ float ssc[NPG_];
    const int g   = blockIdx.x;
    const int tid = threadIdx.x;
    const int n   = g * NPG_ + tid;
    const bool msk = nmask[n] != 0;
    const float sc = msk ? score[n] : NEGV;
    ssc[tid] = sc;
    __syncthreads();
    int rank = 0;
    for (int j = 0; j < NPG_; ++j) {
        const float o = ssc[j];
        rank += (o > sc) || (o == sc && j < tid);
    }
    const bool keep = (rank < k) && msk;
    nmask[n] = keep ? 1 : 0;
    gate[n]  = keep ? tanhf(sc) : 0.f;
}

__global__ void apply_gate(float* __restrict__ h, const float* __restrict__ gate)
{
    const int i = blockIdx.x * blockDim.x + threadIdx.x;   // over N*H/4
    if (i >= NN_ * HH_ / 4) return;
    const float gv = gate[i >> 5];                          // (i*4)>>7
    float4 v = ((float4*)h)[i];
    v.x *= gv; v.y *= gv; v.z *= gv; v.w *= gv;
    ((float4*)h)[i] = v;
}

__global__ void emask_update(unsigned char* __restrict__ emask,
                             const int* __restrict__ src, const int* __restrict__ dst,
                             const unsigned char* __restrict__ nmask)
{
    const int e = blockIdx.x * blockDim.x + threadIdx.x;
    if (e >= EE_) return;
    emask[e] = (emask[e] && nmask[src[e]] && nmask[dst[e]]) ? 1 : 0;
}

// Per-graph sum & max readout -> flat[g, layer*256 + {0..127 | 128..255}]
__global__ void readout_kernel(const float* __restrict__ h,
                               const unsigned char* __restrict__ nmask,
                               float* __restrict__ flat, int layer)
{
    const int g = blockIdx.x;
    const int f = threadIdx.x;       // 128 threads
    float add = 0.f, mx = NEGV;
    const int base = g * NPG_;
    for (int j = 0; j < NPG_; ++j) {
        const int n = base + j;
        const float v = h[(size_t)n * HH_ + f];
        if (nmask[n]) { add += v; mx = fmaxf(mx, v); }
    }
    float* row = flat + (size_t)g * FLATD_ + layer * 256;
    row[f]       = add;
    row[128 + f] = mx;
}

// ---------------------------------------------------------------------------
// Host launcher
// ---------------------------------------------------------------------------
static inline size_t al256(size_t x) { return (x + 255) & ~(size_t)255; }

extern "C" void kernel_launch(void* const* d_in, const int* in_sizes, int n_in,
                              void* d_out, int out_size, void* d_ws, size_t ws_size,
                              hipStream_t stream)
{
    (void)in_sizes; (void)n_in; (void)out_size; (void)ws_size;

    const float* x    = (const float*)d_in[0];
    const int*   ei   = (const int*)d_in[1];
    const int*   srcp = ei;
    const int*   dstp = ei + EE_;
    float*       outp = (float*)d_out;

    // ----- workspace carve-up -----
    char* p = (char*)d_ws;
    auto take = [&](size_t bytes) { char* r = p; p += al256(bytes); return r; };
    float* h     = (float*)take((size_t)NN_ * HH_ * 4);
    float* xl    = (float*)take((size_t)NN_ * HH_ * 4);
    float* xr    = (float*)take((size_t)NN_ * HH_ * 4);
    float* acc   = (float*)take((size_t)NN_ * HH_ * 4);
    float* evals = (float*)take((size_t)(EE_ + NN_) * 4);
    float* emax  = (float*)take((size_t)NN_ * 4);
    float* den   = (float*)take((size_t)NN_ * 4);
    float* score = (float*)take((size_t)NN_ * 4);
    float* gate  = (float*)take((size_t)NN_ * 4);
    float* flat  = (float*)take((size_t)BB_ * FLATD_ * 4);
    float* hid   = (float*)take((size_t)BB_ * HID_ * 4);
    __bf16* aP   = (__bf16*)take((size_t)NN_ * HH_ * 2);     // packed A (reused)
    __bf16* bP   = (__bf16*)take((size_t)FLATD_ * HID_ * 2); // packed B (reused)
    unsigned char* nmask = (unsigned char*)take(NN_);
    unsigned char* emask = (unsigned char*)take(EE_);

    static const int KS[4] = {410, 328, 263, 211};

    const int TPB = 256;
    const int totalEdges = EE_ + NN_;
    const int gridEdgeT  = (totalEdges + TPB - 1) / TPB;
    const int gridEdgeW  = (totalEdges * 32 + TPB - 1) / TPB;
    const int gridNH4    = (NN_ * HH_ / 4 + TPB - 1) / TPB;
    const int gridNodeW  = (NN_ * 32 + TPB - 1) / TPB;
    const int gridE      = (EE_ + TPB - 1) / TPB;

    auto cdiv = [&](int a, int b) { return (a + b - 1) / b; };
    auto gemmGrid = [&](int tilesM, int tilesN, int tnt) {
        return cdiv(tilesM * cdiv(tilesN, tnt), 8);   // 8 waves / 256-thread block
    };

    init_masks<<<gridE, TPB, 0, stream>>>(nmask, emask);

    for (int layer = 0; layer < 4; ++layer) {
        const int pb = 3 + layer * 11;
        const float* Wl    = (const float*)d_in[pb + 0];
        const float* bl    = (const float*)d_in[pb + 1];
        const float* Wr    = (const float*)d_in[pb + 2];
        const float* br    = (const float*)d_in[pb + 3];
        const float* att   = (const float*)d_in[pb + 4];
        const float* cbias = (const float*)d_in[pb + 5];
        const float* gamma = (const float*)d_in[pb + 6];
        const float* beta  = (const float*)d_in[pb + 7];
        const float* mean  = (const float*)d_in[pb + 8];
        const float* var   = (const float*)d_in[pb + 9];
        const float* poolw = (const float*)d_in[pb + 10];

        const float* Ain = (layer == 0) ? x : h;

        pack_a<<<cdiv(NN_ * FIN_ / 8, TPB), TPB, 0, stream>>>(Ain, aP, NN_ * FIN_);

        pack_b_frag<<<cdiv(FIN_ * HH_, TPB), TPB, 0, stream>>>(Wl, bP, FIN_, HH_);
        wmma_gemm<4, 0><<<gemmGrid(NN_ / 16, HH_ / 16, 4), TPB, 0, stream>>>(
            aP, bP, bl, xl, NN_, FIN_, HH_);

        pack_b_frag<<<cdiv(FIN_ * HH_, TPB), TPB, 0, stream>>>(Wr, bP, FIN_, HH_);
        wmma_gemm<4, 0><<<gemmGrid(NN_ / 16, HH_ / 16, 4), TPB, 0, stream>>>(
            aP, bP, br, xr, NN_, FIN_, HH_);

        init_layer<<<gridNH4, TPB, 0, stream>>>(acc, emax, den);
        edge_score<<<gridEdgeW, TPB, 0, stream>>>(xl, xr, att, srcp, dstp, emask, nmask, evals);
        edge_max<<<gridEdgeT, TPB, 0, stream>>>(evals, dstp, emax);
        edge_expsum<<<gridEdgeT, TPB, 0, stream>>>(evals, srcp, dstp, emask, nmask, emax, den);
        edge_scatter<<<gridEdgeW, TPB, 0, stream>>>(evals, srcp, dstp, den, xl, acc);

        bias_bn_relu<<<gridNH4, TPB, 0, stream>>>(acc, cbias, gamma, beta, mean, var, nmask, h);

        score_kernel<<<gridNodeW, TPB, 0, stream>>>(h, poolw, score);
        topk_kernel<<<BB_, NPG_, 0, stream>>>(score, nmask, gate, KS[layer]);
        apply_gate<<<gridNH4, TPB, 0, stream>>>(h, gate);
        emask_update<<<gridE, TPB, 0, stream>>>(emask, srcp, dstp, nmask);

        readout_kernel<<<BB_, HH_, 0, stream>>>(h, nmask, flat, layer);
    }

    const float* W1 = (const float*)d_in[47];
    const float* b1 = (const float*)d_in[48];
    const float* W2 = (const float*)d_in[49];
    const float* b2 = (const float*)d_in[50];

    // dense1: (128 x 1024) @ (1024 x 512) + b, relu
    pack_a<<<cdiv(BB_ * FLATD_ / 8, TPB), TPB, 0, stream>>>(flat, aP, BB_ * FLATD_);
    pack_b_frag<<<cdiv(FLATD_ * HID_, TPB), TPB, 0, stream>>>(W1, bP, FLATD_, HID_);
    wmma_gemm<4, 1><<<gemmGrid(BB_ / 16, HID_ / 16, 4), TPB, 0, stream>>>(
        aP, bP, b1, hid, BB_, FLATD_, HID_);

    // dense2: (128 x 512) @ (512 x 10) + b
    pack_a<<<cdiv(BB_ * HID_ / 8, TPB), TPB, 0, stream>>>(hid, aP, BB_ * HID_);
    pack_b_frag<<<cdiv(HID_ * CC_, TPB), TPB, 0, stream>>>(W2, bP, HID_, CC_);
    wmma_gemm<1, 0><<<gemmGrid(BB_ / 16, 1, 1), TPB, 0, stream>>>(
        aP, bP, b2, outp, BB_, HID_, CC_);
}
